// HashRoutedMoE_7378753815192
// MI455X (gfx1250) — matrix operand order repaired and checked by reading, compiled, not run on hardware
//
#include <hip/hip_runtime.h>
#include <hip/hip_bf16.h>
#include <math.h>

// ---------------- problem constants ----------------
#define BB 8
#define TT 4096
#define DM 512
#define DF 2048
#define NE 8
#define NTOK (BB * TT)            // 32768 tokens
#define CAP NTOK                  // worst-case tokens per expert list
#define MTILE 32                  // tokens per block tile
#define MAX_TILES (CAP / MTILE)   // 1024 tiles per expert
#define OUT_ELEMS ((size_t)NTOK * DM)
#define IDS_ELEMS ((size_t)NTOK * 2)
#define WELEMS ((size_t)NE * DM * DF)   // per weight tensor: 8.39M elems
#define XELEMS ((size_t)NTOK * DM)      // x: 16.8M elems

#if __has_builtin(__builtin_amdgcn_global_load_async_to_lds_b128) && \
    __has_builtin(__builtin_amdgcn_s_wait_asynccnt)
#define HAVE_ASYNC_LDS 1
#else
#define HAVE_ASYNC_LDS 0
#endif

// ---------------- WMMA types ----------------
typedef __attribute__((ext_vector_type(16))) __bf16 v16bf;
typedef __attribute__((ext_vector_type(8)))  float  v8f;

#if HAVE_ASYNC_LDS
// exact parameter types from the builtin's signature (per hipcc diagnostic):
// (v4i addrspace(1)*, v4i addrspace(3)*, imm int offset, imm int cpol)
typedef int v4i_async __attribute__((vector_size(16)));
typedef __attribute__((address_space(1))) v4i_async* async_gptr;
typedef __attribute__((address_space(3))) v4i_async* async_lptr;
#endif

union ABu { unsigned u[8]; uint4 q[2]; v16bf v; };

// round-to-nearest-even f32 -> bf16 (raw ushort bits)
static __device__ inline unsigned short f2bf(float f) {
    unsigned u = __float_as_uint(f);
    unsigned r = ((u >> 16) & 1u) + 0x7fffu;
    return (unsigned short)((u + r) >> 16);
}

// A fragment (16x32 bf16) from an LDS row (row-major bf16, 16B-aligned rows).
static __device__ inline v16bf make_A(const unsigned* row, int k0, int lhi) {
    ABu t;
    int idx = (k0 >> 1) + 8 * lhi;            // dword index, multiple of 4
    t.q[0] = *(const uint4*)(row + idx);      // ds_load_b128
    t.q[1] = *(const uint4*)(row + idx + 4);  // ds_load_b128
    return t.v;
}

// B fragment (32x16 bf16) from K-major pre-converted bf16 weights.
static __device__ inline v16bf make_B(const unsigned short* col_k) {
    ABu t;
    const uint4* p = (const uint4*)col_k;     // 32B aligned
    t.q[0] = p[0];                            // global_load_b128
    t.q[1] = p[1];                            // global_load_b128
    return t.v;
}

static __device__ inline v8f wmma_bf16(v16bf a, v16bf b, v8f c) {
    return __builtin_amdgcn_wmma_f32_16x16x32_bf16(
        false, a, false, b, (short)0, c, false, false);
}

// ---------------- kernel 0: zero out buffer + counters ----------------
__global__ void moe_zero_kernel(float* __restrict__ out, int* __restrict__ cnt) {
    if (blockIdx.x == 0 && threadIdx.x < NE) cnt[threadIdx.x] = 0;
    size_t stride = (size_t)gridDim.x * blockDim.x;
    for (size_t i = (size_t)blockIdx.x * blockDim.x + threadIdx.x; i < OUT_ELEMS; i += stride)
        out[i] = 0.0f;
}

// ---------------- kernel 1: precision/layout conversion ----------------
// Wt1[e][ff][dm] <- W1[e][dm][ff];  Wt2[e][dm][ff] <- W2[e][ff][dm];  xbf <- bf16(x)
__global__ void moe_cvt_kernel(const float* __restrict__ W1, const float* __restrict__ W2,
                               const float* __restrict__ x,
                               unsigned short* __restrict__ Wt1, unsigned short* __restrict__ Wt2,
                               unsigned short* __restrict__ xbf) {
    size_t stride = (size_t)gridDim.x * blockDim.x;
    for (size_t i = (size_t)blockIdx.x * blockDim.x + threadIdx.x; i < WELEMS; i += stride) {
        {   // Wt1: i = ((e*DF + n)*DM + k)
            size_t e = i / ((size_t)DF * DM);
            size_t r = i % ((size_t)DF * DM);
            size_t n = r / DM, k = r % DM;
            Wt1[i] = f2bf(W1[e * ((size_t)DM * DF) + k * DF + n]);
        }
        {   // Wt2: i = ((e*DM + n)*DF + k)
            size_t e = i / ((size_t)DM * DF);
            size_t r = i % ((size_t)DM * DF);
            size_t n = r / DF, k = r % DF;
            Wt2[i] = f2bf(W2[e * ((size_t)DF * DM) + k * DM + n]);
        }
    }
    for (size_t i = (size_t)blockIdx.x * blockDim.x + threadIdx.x; i < XELEMS; i += stride)
        xbf[i] = f2bf(x[i]);
}

// ---------------- kernel 2: hash routing + gather lists (block-aggregated) ----------------
__global__ void moe_route_kernel(const int* __restrict__ token_ids,
                                 int* __restrict__ ids_out,
                                 int* __restrict__ cnt,
                                 int* __restrict__ lists) {
    __shared__ int lc[NE];
    __shared__ int base[NE];
    if (threadIdx.x < NE) lc[threadIdx.x] = 0;
    __syncthreads();

    int t = blockIdx.x * blockDim.x + threadIdx.x;
    int e0 = 0, e1 = 0, p0 = 0, p1 = 0;
    bool valid = t < NTOK;
    if (valid) {
        int b = t / TT;
        e0 = ((b * TT) + token_ids[t]) & (NE - 1);
        e1 = (e0 + 1) & (NE - 1);
        ids_out[2 * t + 0] = e0;
        ids_out[2 * t + 1] = e1;
        p0 = atomicAdd(&lc[e0], 1);
        p1 = atomicAdd(&lc[e1], 1);
    }
    __syncthreads();
    if (threadIdx.x < NE)
        base[threadIdx.x] = atomicAdd(&cnt[threadIdx.x], lc[threadIdx.x]);
    __syncthreads();
    if (valid) {
        lists[e0 * CAP + base[e0] + p0] = t;
        lists[e1 * CAP + base[e1] + p1] = t;
    }
}

// ---------------- kernel 3: grouped expert MLP via bf16 WMMA, 32-token tiles ----------------
__global__ __launch_bounds__(256) void moe_tile_kernel(
    const unsigned short* __restrict__ xbf,
    const unsigned short* __restrict__ Wt1, const float* __restrict__ b1,
    const unsigned short* __restrict__ Wt2, const float* __restrict__ b2,
    float* __restrict__ out, float* __restrict__ counts_out,
    const int* __restrict__ cnt, const int* __restrict__ lists) {

    __shared__ unsigned short Xs[MTILE][DM];    // 32 KB
    __shared__ unsigned short Hs[MTILE][1024];  // 64 KB (half of d_ff per phase)
    __shared__ int tok[MTILE];

    const int e     = blockIdx.x / MAX_TILES;
    const int tile  = blockIdx.x % MAX_TILES;
    const int cnt_e = cnt[e];

    if (tile == 0 && threadIdx.x == 0) counts_out[e] = (float)cnt_e;
    if (tile * MTILE >= cnt_e) return;

    if (threadIdx.x < MTILE) {
        int r = tile * MTILE + threadIdx.x;
        tok[threadIdx.x] = (r < cnt_e) ? lists[e * CAP + r] : -1;
    }
    __syncthreads();

    // ---- stage X tile (bf16 rows, 16B chunks) ----
    for (int i = threadIdx.x; i < MTILE * (DM / 8); i += 256) {
        int r = i >> 6, c4 = i & 63;          // DM/8 = 64 uint4 per row
        int t = tok[r];
        if (t >= 0) {
#if HAVE_ASYNC_LDS
            // per-lane async gather: global -> LDS, tracked by ASYNCcnt
            __builtin_amdgcn_global_load_async_to_lds_b128(
                (async_gptr)(xbf + (size_t)t * DM + c4 * 8),
                (async_lptr)&Xs[r][c4 * 8],
                0, 0);
#else
            ((uint4*)&Xs[r][0])[c4] = ((const uint4*)(xbf + (size_t)t * DM))[c4];
#endif
        } else {
            ((uint4*)&Xs[r][0])[c4] = make_uint4(0u, 0u, 0u, 0u);
        }
    }
#if HAVE_ASYNC_LDS
    __builtin_amdgcn_s_wait_asynccnt(0);
#endif
    __syncthreads();

    const int wave = threadIdx.x >> 5;
    const int lane = threadIdx.x & 31;
    const int lm   = lane & 15;
    const int lhi  = lane >> 4;

    const unsigned short* Wt1e = Wt1 + (size_t)e * DF * DM;
    const unsigned short* Wt2e = Wt2 + (size_t)e * DM * DF;
    const float* b1e = b1 + (size_t)e * DF;
    const float* b2e = b2 + (size_t)e * DM;

    const unsigned* XsRow0 = (const unsigned*)&Xs[lm][0];
    const unsigned* XsRow1 = (const unsigned*)&Xs[16 + lm][0];
    const unsigned* HsRow0 = (const unsigned*)&Hs[lm][0];
    const unsigned* HsRow1 = (const unsigned*)&Hs[16 + lm][0];

    v8f acc2[2][4];
#pragma unroll
    for (int rt = 0; rt < 2; ++rt)
#pragma unroll
        for (int s = 0; s < 4; ++s) acc2[rt][s] = (v8f){};

    for (int phase = 0; phase < 2; ++phase) {
        const int ffbase = phase * 1024;

        // ---- GEMM1: k-outer; 2 A fragments x 8 B columns -> 16 WMMA per slab ----
        v8f acc1[2][8];
#pragma unroll
        for (int rt = 0; rt < 2; ++rt)
#pragma unroll
            for (int s = 0; s < 8; ++s) acc1[rt][s] = (v8f){};

        for (int k0 = 0; k0 < DM; k0 += 32) {
            v16bf A0 = make_A(XsRow0, k0, lhi);
            v16bf A1 = make_A(XsRow1, k0, lhi);
            __builtin_prefetch(Wt1e + (size_t)(ffbase + wave * 128 + lm) * DM
                                    + (k0 + 32) + 16 * lhi, 0, 0);
#pragma unroll
            for (int sub = 0; sub < 8; ++sub) {
                const int ffcol = ffbase + wave * 128 + sub * 16 + lm;
                v16bf B = make_B(Wt1e + (size_t)ffcol * DM + k0 + 16 * lhi);
                acc1[0][sub] = wmma_bf16(A0, B, acc1[0][sub]);
                acc1[1][sub] = wmma_bf16(A1, B, acc1[1][sub]);
            }
        }

        // bias + exact GELU + store H tile (bf16) to LDS
#pragma unroll
        for (int sub = 0; sub < 8; ++sub) {
            const int nH = wave * 128 + sub * 16 + lm;
            const float bias = b1e[ffbase + nH];
#pragma unroll
            for (int rt = 0; rt < 2; ++rt)
#pragma unroll
                for (int i = 0; i < 8; ++i) {
                    float v = acc1[rt][sub][i] + bias;
                    v = 0.5f * v * (1.0f + erff(v * 0.70710678118654752f));
                    Hs[rt * 16 + i + 8 * lhi][nH] = f2bf(v);
                }
        }
        __syncthreads();

        // ---- GEMM2 partial: 2 A fragments x 4 B columns -> 8 WMMA per slab ----
        for (int k0 = 0; k0 < 1024; k0 += 32) {
            v16bf A0 = make_A(HsRow0, k0, lhi);
            v16bf A1 = make_A(HsRow1, k0, lhi);
#pragma unroll
            for (int sub = 0; sub < 4; ++sub) {
                const int dcol = wave * 64 + sub * 16 + lm;
                v16bf B = make_B(Wt2e + (size_t)dcol * DF + (ffbase + k0) + 16 * lhi);
                acc2[0][sub] = wmma_bf16(A0, B, acc2[0][sub]);
                acc2[1][sub] = wmma_bf16(A1, B, acc2[1][sub]);
            }
        }
        __syncthreads();  // before next phase overwrites Hs
    }

    // ---- epilogue: bias, combine weight 1/4, atomic accumulate ----
#pragma unroll
    for (int sub = 0; sub < 4; ++sub) {
        const int dcol = wave * 64 + sub * 16 + lm;
        const float bias = b2e[dcol];
#pragma unroll
        for (int rt = 0; rt < 2; ++rt)
#pragma unroll
            for (int i = 0; i < 8; ++i) {
                int m = rt * 16 + i + 8 * lhi;
                int t = tok[m];
                if (t >= 0)
                    atomicAdd(&out[(size_t)t * DM + dcol], (acc2[rt][sub][i] + bias) * 0.25f);
            }
    }
}

// ---------------- host launcher ----------------
extern "C" void kernel_launch(void* const* d_in, const int* in_sizes, int n_in,
                              void* d_out, int out_size, void* d_ws, size_t ws_size,
                              hipStream_t stream) {
    const float* x         = (const float*)d_in[0];
    const int*   token_ids = (const int*)d_in[1];
    const float* W1        = (const float*)d_in[2];
    const float* b1        = (const float*)d_in[3];
    const float* W2        = (const float*)d_in[4];
    const float* b2        = (const float*)d_in[5];

    float* out        = (float*)d_out;
    int*   ids_out    = (int*)((float*)d_out + OUT_ELEMS);
    float* counts_out = (float*)d_out + OUT_ELEMS + IDS_ELEMS;

    // workspace: [cnt 64B][lists 1MB][Wt1 16.8MB][Wt2 16.8MB][xbf 33.6MB]
    int* cnt   = (int*)d_ws;
    int* lists = cnt + 16;
    unsigned short* Wt1 = (unsigned short*)(lists + (size_t)NE * CAP);
    unsigned short* Wt2 = Wt1 + WELEMS;
    unsigned short* xbf = Wt2 + WELEMS;

    moe_zero_kernel<<<4096, 256, 0, stream>>>(out, cnt);
    moe_cvt_kernel<<<8192, 256, 0, stream>>>(W1, W2, x, Wt1, Wt2, xbf);
    moe_route_kernel<<<(NTOK + 255) / 256, 256, 0, stream>>>(token_ids, ids_out, cnt, lists);
    moe_tile_kernel<<<NE * MAX_TILES, 256, 0, stream>>>(xbf, Wt1, b1, Wt2, b2,
                                                        out, counts_out, cnt, lists);
}
